// GNNDualLayer_89215060672585
// MI455X (gfx1250) — matrix operand order repaired and checked by reading, not compiled, not run on hardware
//
#include <hip/hip_runtime.h>

// ---------------------------------------------------------------------------
// MI455X (gfx1250) GNN dual-layer.
// Roofline: ~20 MB mandatory traffic (dominated by the two int32 adjacency
// matrices) => ~0.86 us at 23.3 TB/s HBM. Compute (~0.45 GFLOP) is noise for
// the WMMA units, so: read adjacency exactly once, serve features from
// L2/LDS (via the Tensor Data Mover where it fits), and run all matmuls
// through v_wmma_f32_16x16x32_f16 (f16 in / f32 accumulate; adjacency 0/1 is
// exact in f16, feature rounding ~5e-4 rel).
// ---------------------------------------------------------------------------

#define NN 1536          // N1 == N2
#define DD 64            // D_IN == D_OUT
#define NF (NN * DD)     // 98304 elements per feature matrix

typedef __attribute__((ext_vector_type(16))) _Float16 v16h;
typedef __attribute__((ext_vector_type(8)))  _Float16 v8h;
typedef __attribute__((ext_vector_type(8)))  float    v8f;

#if defined(__HIP_DEVICE_COMPILE__) && __has_builtin(__builtin_amdgcn_tensor_load_to_lds)
#define HAVE_TDM 1
#else
#define HAVE_TDM 0
#endif

// ---- WMMA fragment loaders (layouts per CDNA5 ISA 7.12.2, wave32) ----------
// A 16x32 f16, row-major source, leading dim ld (halves):
//   lane t: row = base+(t&15), h = t>>4
//   elems 0..7  -> K = kb + 8h + e         (8 contiguous halves)
//   elems 8..15 -> K = kb + 16 + 8h + (e-8)(8 contiguous halves)
__device__ __forceinline__ v16h load_a_f16(const _Float16* __restrict__ X,
                                           int row, int kb, int ld) {
  const int lane = threadIdx.x & 31;
  const int h    = lane >> 4;
  const _Float16* p = X + (size_t)row * ld + kb + 8 * h;
  v8h lo = *(const v8h*)(p);
  v8h hi = *(const v8h*)(p + 16);
  return __builtin_shufflevector(lo, hi, 0,1,2,3,4,5,6,7,8,9,10,11,12,13,14,15);
}

// A 16x32 from an int32 0/1 adjacency row (convert to f16 in registers).
__device__ __forceinline__ v16h load_a_adj(const int* __restrict__ adj,
                                           int row, int kb, int ldk) {
  const int lane = threadIdx.x & 31;
  const int h    = lane >> 4;
  const int* p = adj + (size_t)row * ldk + kb + 8 * h;
  int4 a0 = *(const int4*)(p);
  int4 a1 = *(const int4*)(p + 4);
  int4 b0 = *(const int4*)(p + 16);
  int4 b1 = *(const int4*)(p + 20);
  v16h a;
  a[0]=(_Float16)a0.x; a[1]=(_Float16)a0.y; a[2]=(_Float16)a0.z; a[3]=(_Float16)a0.w;
  a[4]=(_Float16)a1.x; a[5]=(_Float16)a1.y; a[6]=(_Float16)a1.z; a[7]=(_Float16)a1.w;
  a[8]=(_Float16)b0.x; a[9]=(_Float16)b0.y; a[10]=(_Float16)b0.z; a[11]=(_Float16)b0.w;
  a[12]=(_Float16)b1.x; a[13]=(_Float16)b1.y; a[14]=(_Float16)b1.z; a[15]=(_Float16)b1.w;
  return a;
}

// B 32x16, stored K-major (Bk[col][k] contiguous in k), leading dim ld:
//   lane t: col = base+(t&15), h = t>>4, elem e<8 -> K = kb + 8h + e, e>=8 -> K = kb + 16 + 8h + (e-8)
__device__ __forceinline__ v16h load_b_kmajor(const _Float16* __restrict__ B,
                                              int col, int kb, int ld) {
  const int lane = threadIdx.x & 31;
  const int h    = lane >> 4;
  const _Float16* p = B + (size_t)col * ld + kb + 8 * h;
  v8h lo = *(const v8h*)(p);
  v8h hi = *(const v8h*)(p + 16);
  return __builtin_shufflevector(lo, hi, 0,1,2,3,4,5,6,7,8,9,10,11,12,13,14,15);
}

__device__ __forceinline__ v8f wmma16(v16h a, v16h b, v8f c) {
  return __builtin_amdgcn_wmma_f32_16x16x32_f16(
      /*neg_a=*/false, a, /*neg_b=*/false, b,
      /*c_mod=*/(short)0, c, /*reuse_a=*/false, /*reuse_b=*/false);
}

// ---- Tensor Data Mover: async DMA of one 32x64 f32 tile into LDS ----------
// D# per CDNA5 ISA ch.8: group0 = {count, lds_addr, global_addr, type=2},
// group1 = {data_size=4B, tensor 64x1536, tile 64x32, dim0_stride=64}.
#if HAVE_TDM
typedef __attribute__((ext_vector_type(4))) unsigned int u32x4;
typedef __attribute__((ext_vector_type(4))) int          i32x4;
typedef __attribute__((ext_vector_type(8))) int          i32x8;

__device__ __forceinline__ void tdm_feats_tile(const float* base, int jrow,
                                               unsigned lds_byte_off) {
  unsigned long long ga =
      (unsigned long long)(const void*)(base + (size_t)jrow * DD);
  u32x4 g0;
  g0[0] = 1u;                                      // count=1, user descriptor
  g0[1] = lds_byte_off;                            // LDS dest (bytes)
  g0[2] = (unsigned)(ga & 0xFFFFFFFFu);            // global_addr[31:0]
  g0[3] = (unsigned)((ga >> 32) & 0x01FFFFFFu)     // global_addr[56:32]
        | (2u << 30);                              // type = 2 ("image")
  i32x8 g1;
  g1[0] = 0x00020000;                              // data_size = 2 -> 4 bytes
  g1[1] = (DD & 0xFFFF) << 16;                     // tensor_dim0 = 64
  g1[2] = (NN & 0xFFFF) << 16;                     // tensor_dim1 = 1536
  g1[3] = (DD & 0xFFFF) << 16;                     // tile_dim0  = 64
  g1[4] = 32;                                      // tile_dim1  = 32
  g1[5] = DD;                                      // tensor_dim0_stride = 64
  g1[6] = 0;
  g1[7] = 0;
  i32x4 z4 = {0, 0, 0, 0};                         // 2-D tensor: groups 2/3 off
#if __clang_major__ >= 23
  i32x8 z8 = {0, 0, 0, 0, 0, 0, 0, 0};
  __builtin_amdgcn_tensor_load_to_lds(g0, g1, z4, z4, z8, 0);
#else
  __builtin_amdgcn_tensor_load_to_lds(g0, g1, z4, z4, 0);
#endif
}
#endif  // HAVE_TDM

// ---- K0: convert features + weights to f16 (and K-major transpose of f1) --
__global__ void k_convert(const float* __restrict__ f1, const float* __restrict__ f2,
                          const float* __restrict__ w1s, const float* __restrict__ w1n,
                          const float* __restrict__ w2s, const float* __restrict__ w2n,
                          _Float16* __restrict__ f1_h, _Float16* __restrict__ f1T_h,
                          _Float16* __restrict__ f2_h,
                          _Float16* __restrict__ w1s_h, _Float16* __restrict__ w1n_h,
                          _Float16* __restrict__ w2s_h, _Float16* __restrict__ w2n_h) {
  const int i = blockIdx.x * 256 + threadIdx.x;   // grid covers NF exactly
  const float a = f1[i];
  f1_h[i] = (_Float16)a;
  const int row = i >> 6, col = i & 63;
  f1T_h[(size_t)col * NN + row] = (_Float16)a;    // K-major copy for GEMM B
  f2_h[i] = (_Float16)f2[i];
  if (i < DD * DD) {
    w1s_h[i] = (_Float16)w1s[i]; w1n_h[i] = (_Float16)w1n[i];
    w2s_h[i] = (_Float16)w2s[i]; w2n_h[i] = (_Float16)w2n[i];
  }
}

// ---- K1: masked max aggregation (agg1 = maxmask(adj_2to1, feats2)) ---------
// Block = 256 threads, 32 rows x 64 feats; thread t: row = t>>3, 8 feats.
// feats2 tiles stream through LDS via double-buffered TDM; adjacency tiles
// arrive via normal vector loads so both engines overlap.
__global__ void k_maxagg(const int* __restrict__ adj,      // [NN][NN]
                         const float* __restrict__ feats2, // [NN][DD]
                         _Float16* __restrict__ agg1_h) {  // [NN][DD]
  __shared__ struct {
    float f[2][32][DD];   // 2 x 8KB TDM buffers (struct offset 0)
    int   adj[32][32];    // 4KB adjacency tile
  } sm;
  const int t  = threadIdx.x;
  const int r  = t >> 3;
  const int fb = (t & 7) * 8;
  const int rowBase = blockIdx.x * 32;

  float acc[8];
#pragma unroll
  for (int k = 0; k < 8; ++k) acc[k] = -3.402823466e38f;
  int has = 0;

  const int T = NN / 32;   // 48 tiles
#if HAVE_TDM
  if (t == 0) tdm_feats_tile(feats2, 0, 0u);   // preload tile 0 into buf 0
#endif

  for (int i = 0; i < T; ++i) {
    const int cur = i & 1;
    const int jb  = i * 32;
#if HAVE_TDM
    if (t == 0 && i + 1 < T)                    // start next tile early
      tdm_feats_tile(feats2, jb + 32, (unsigned)((cur ^ 1) * 32 * DD * 4));
#else
    {
      const float* src = feats2 + (size_t)(jb + r) * DD + fb;
      *(float4*)&sm.f[cur][r][fb]     = *(const float4*)(src);
      *(float4*)&sm.f[cur][r][fb + 4] = *(const float4*)(src + 4);
    }
#endif
    {
      const int lc = (t & 7) * 4;
      *(int4*)&sm.adj[r][lc] =
          *(const int4*)(adj + (size_t)(rowBase + r) * NN + jb + lc);
    }
#if HAVE_TDM
    if (t == 0) {
      if (i + 1 < T) __builtin_amdgcn_s_wait_tensorcnt(1);  // oldest done
      else           __builtin_amdgcn_s_wait_tensorcnt(0);
    }
#endif
    __syncthreads();   // adj tile visible + feats tile `cur` DMA complete
#pragma unroll 4
    for (int jj = 0; jj < 32; ++jj) {
      const int a = sm.adj[r][jj];
      has |= a;
      const float4 v0 = *(const float4*)&sm.f[cur][jj][fb];
      const float4 v1 = *(const float4*)&sm.f[cur][jj][fb + 4];
      const float neg = -3.402823466e38f;
      acc[0] = fmaxf(acc[0], a ? v0.x : neg);
      acc[1] = fmaxf(acc[1], a ? v0.y : neg);
      acc[2] = fmaxf(acc[2], a ? v0.z : neg);
      acc[3] = fmaxf(acc[3], a ? v0.w : neg);
      acc[4] = fmaxf(acc[4], a ? v1.x : neg);
      acc[5] = fmaxf(acc[5], a ? v1.y : neg);
      acc[6] = fmaxf(acc[6], a ? v1.z : neg);
      acc[7] = fmaxf(acc[7], a ? v1.w : neg);
    }
    __syncthreads();   // all readers done before buffers are overwritten
  }
  v8h o;
#pragma unroll
  for (int k = 0; k < 8; ++k) o[k] = (_Float16)(has ? acc[k] : 0.0f);
  *(v8h*)(agg1_h + (size_t)(rowBase + r) * DD + fb) = o;
}

// ---- K2: agg2 = adj_1to2 @ feats1 via WMMA (M=1536,K=1536,N=64) ------------
// 8 waves/block; wave = 16x16 tile; block = 32 rows x 64 cols; 48 blocks.
__global__ void k_agg2(const int* __restrict__ adj12,      // [NN][NN]
                       const _Float16* __restrict__ f1T,   // [DD][NN] K-major
                       _Float16* __restrict__ agg2_h) {    // [NN][DD]
  const int wave = threadIdx.x >> 5;
  const int rowBase = blockIdx.x * 32 + (wave >> 2) * 16;
  const int colBase = (wave & 3) * 16;
  const int lane = threadIdx.x & 31;
  const int row = rowBase + (lane & 15);
  const int col = colBase + (lane & 15);
  v8f c = {};
  for (int kb = 0; kb < NN; kb += 32) {
    v16h a = load_a_adj(adj12, row, kb, NN);
    v16h b = load_b_kmajor(f1T, col, kb, NN);
    c = wmma16(a, b, c);
  }
  const int m0 = 8 * (lane >> 4);
#pragma unroll
  for (int r = 0; r < 8; ++r)
    agg2_h[(size_t)(rowBase + m0 + r) * DD + col] = (_Float16)c[r];
}

// ---- K3: out = relu(X @ Ws^T + G @ Wn^T), all f16 in / f32 out -------------
// W stored row-major W[n][k] is already K-major for the B fragment.
__global__ void k_proj(const _Float16* __restrict__ X,   // [NN][DD]
                       const _Float16* __restrict__ G,   // [NN][DD]
                       const _Float16* __restrict__ Ws,  // [DD][DD]
                       const _Float16* __restrict__ Wn,  // [DD][DD]
                       float* __restrict__ out) {        // [NN][DD]
  const int wave = threadIdx.x >> 5;
  const int rowBase = blockIdx.x * 32 + (wave >> 2) * 16;
  const int colBase = (wave & 3) * 16;
  const int lane = threadIdx.x & 31;
  const int row = rowBase + (lane & 15);
  const int col = colBase + (lane & 15);
  v8f c = {};
#pragma unroll
  for (int kb = 0; kb < DD; kb += 32) {
    v16h ax = load_a_f16(X, row, kb, DD);
    v16h bs = load_b_kmajor(Ws, col, kb, DD);
    c = wmma16(ax, bs, c);
    v16h ag = load_a_f16(G, row, kb, DD);
    v16h bn = load_b_kmajor(Wn, col, kb, DD);
    c = wmma16(ag, bn, c);
  }
  const int m0 = 8 * (lane >> 4);
#pragma unroll
  for (int r = 0; r < 8; ++r)
    out[(size_t)(rowBase + m0 + r) * DD + col] = fmaxf(c[r], 0.0f);
}

// ---------------------------------------------------------------------------
extern "C" void kernel_launch(void* const* d_in, const int* in_sizes, int n_in,
                              void* d_out, int out_size, void* d_ws, size_t ws_size,
                              hipStream_t stream) {
  const float* node_feats1 = (const float*)d_in[0];
  const float* node_feats2 = (const float*)d_in[1];
  const int*   adj_1to2    = (const int*)d_in[2];
  const int*   adj_2to1    = (const int*)d_in[3];
  const float* w1_self     = (const float*)d_in[4];
  const float* w1_neigh    = (const float*)d_in[5];
  const float* w2_self     = (const float*)d_in[6];
  const float* w2_neigh    = (const float*)d_in[7];

  float* out1 = (float*)d_out;
  float* out2 = out1 + NF;

  // workspace carve-up (all f16): ~0.97 MB total
  _Float16* ws = (_Float16*)d_ws;
  _Float16* f1_h   = ws;                 // [NN][DD]
  _Float16* f1T_h  = f1_h   + NF;        // [DD][NN] K-major
  _Float16* f2_h   = f1T_h  + NF;        // [NN][DD]
  _Float16* agg1_h = f2_h   + NF;        // [NN][DD]
  _Float16* agg2_h = agg1_h + NF;        // [NN][DD]
  _Float16* w1s_h  = agg2_h + NF;        // [DD][DD]
  _Float16* w1n_h  = w1s_h  + DD * DD;
  _Float16* w2s_h  = w1n_h  + DD * DD;
  _Float16* w2n_h  = w2s_h  + DD * DD;

  k_convert<<<NF / 256, 256, 0, stream>>>(
      node_feats1, node_feats2, w1_self, w1_neigh, w2_self, w2_neigh,
      f1_h, f1T_h, f2_h, w1s_h, w1n_h, w2s_h, w2n_h);

  k_maxagg<<<NN / 32, 256, 0, stream>>>(adj_2to1, node_feats2, agg1_h);

  k_agg2<<<NN / 32, 256, 0, stream>>>(adj_1to2, f1T_h, agg2_h);

  k_proj<<<NN / 32, 256, 0, stream>>>(f1_h, agg1_h, w1s_h, w1n_h, out1);
  k_proj<<<NN / 32, 256, 0, stream>>>(f2_h, agg2_h, w2s_h, w2n_h, out2);
}
